// Embedding_48326972015371
// MI455X (gfx1250) — matrix operand order repaired and checked by reading, not compiled, not run on hardware
//
#include <hip/hip_runtime.h>
#include <hip/hip_bf16.h>

typedef _Float16 v8h  __attribute__((ext_vector_type(8)));
typedef _Float16 v16h __attribute__((ext_vector_type(16)));
typedef float    v8f  __attribute__((ext_vector_type(8)));
typedef unsigned int u32x4 __attribute__((ext_vector_type(4)));
typedef int          i32x8 __attribute__((ext_vector_type(8)));
typedef int          i32x4 __attribute__((ext_vector_type(4)));

#define NPTS 65536
#define CIN  5
#define DD   128
#define KNB  15
#define HH   64
#define WW   1024
#define HWSZ 65536
#define TPB  256

// A-fragment (16x32 f16, M x K) for v_wmma_f32_16x16x32_f16.
// Lanes 0-15: row M=lane, K+0..15;
// lanes 16-31: same row, K+16..31.
static __device__ __forceinline__ v16h load_a_frag(const _Float16* p) {
    union { v16h v; v8h h[2]; } u;
    u.h[0] = *(const v8h*)(p);
    u.h[1] = *(const v8h*)(p + 8);
    return u.v;
}

// Branch-free exact GELU: 0.5*a*(1+erf(a/sqrt2)) with A&S 7.1.26 erf
// (max abs err 1.5e-7, far below the f16 quantization applied right after).
// Uses v_rcp_f32 + v_exp_f32; no divergent branches.
static __device__ __forceinline__ float gelu_exact(float a) {
    float z  = a * 0.70710678118f;
    float az = fabsf(z);
    float t  = __builtin_amdgcn_rcpf(fmaf(0.3275911f, az, 1.0f));
    float p  = fmaf(1.061405429f, t, -1.453152027f);
    p = fmaf(p, t, 1.421413741f);
    p = fmaf(p, t, -0.284496736f);
    p = fmaf(p, t, 0.254829592f);
    p = p * t;
    float ex = __expf(-az * az);
    float er = fmaf(-p, ex, 1.0f);          // erf(|z|)
    er = copysignf(er, a);                  // erf(z)
    return 0.5f * a * (1.0f + er);
}

// ---------------------------------------------------------------------------
__global__ void init_out_kernel(float* __restrict__ out, int n) {
    int i = blockIdx.x * blockDim.x + threadIdx.x;
    if (i < n) out[i] = -1.0f;
}

// ---------------------------------------------------------------------------
__global__ void prep_kernel(const float* __restrict__ pc,
                            const float* __restrict__ ng, const float* __restrict__ nb,
                            const float* __restrict__ nm, const float* __restrict__ nv,
                            const float* __restrict__ w2b, const float* __restrict__ fw,
                            float* __restrict__ xout,
                            _Float16* __restrict__ w2bh, _Float16* __restrict__ fwh) {
    int i = blockIdx.x * blockDim.x + threadIdx.x;
    if (i < NPTS) {
#pragma unroll
        for (int c = 0; c < CIN; ++c) {
            float s = ng[c] * rsqrtf(nv[c] + 1e-5f);
            xout[i * CIN + c] = (pc[i * CIN + c] - nm[c]) * s + nb[c];
        }
    }
    if (i < DD * DD)     w2bh[i] = (_Float16)w2b[i];
    if (i < DD * 2 * DD) fwh[i]  = (_Float16)fw[i];
}

// ---------------------------------------------------------------------------
// Fused per-16-point tile.
// ---------------------------------------------------------------------------
__global__ __launch_bounds__(TPB) void fused_embed_kernel(
    const float* __restrict__ xg,        // N x 5 BN'd points
    const int*   __restrict__ neighbors, // N x 16
    const int*   __restrict__ px, const int* __restrict__ py,
    const float* __restrict__ c1w, const float* __restrict__ c1b,
    const float* __restrict__ g2a, const float* __restrict__ b2a,
    const float* __restrict__ m2a, const float* __restrict__ v2a,
    const float* __restrict__ w2a,
    const float* __restrict__ g2b, const float* __restrict__ b2b,
    const float* __restrict__ m2b, const float* __restrict__ v2b,
    const _Float16* __restrict__ w2bh,   // 128x128 f16
    const _Float16* __restrict__ fwh,    // 128x256 f16
    const float* __restrict__ fb,
    float* __restrict__ out)             // 128 x 64 x 1024
{
    __shared__ float    sh_w2a[DD * CIN];
    __shared__ float    sh_c1w[DD * CIN];
    __shared__ float    sh_c1b[DD];
    __shared__ float    sh_s2b[DD];
    __shared__ float    sh_o2b[DD];
    __shared__ float    sh_fb[DD];
    __shared__ float    sh_s2a[8];
    __shared__ float    sh_o2a[8];
    __shared__ float    sh_xc[16 * CIN];
    __shared__ float    sh_tall[16 * KNB * CIN];    // BN2a'd diffs, col = k*16+p
    __shared__ int      sh_pix[16];
    __shared__ int      sh_nbr[16 * 16];            // TDM-loaded neighbor tile
    __shared__ _Float16 sh_h1[2][16 * DD];          // double-buffered gelu(h) tile
    __shared__ _Float16 sh_cat[16 * 2 * DD];        // concat columns

    const int tid  = threadIdx.x;
    const int base = blockIdx.x * 16;

    // ---- TDM: async-load the 16x16 int32 neighbor tile into LDS ----
    if (tid < 32) {
        unsigned long long ga  = (unsigned long long)(uintptr_t)(neighbors + (size_t)base * 16);
        unsigned int       lds = (unsigned int)(uintptr_t)(&sh_nbr[0]);   // low 32b = LDS offset
        // D# group0: count=1 | lds_addr | global_addr[56:0] | type=2
        u32x4 g0 = { 1u, lds, (unsigned int)(ga & 0xFFFFFFFFu),
                     (unsigned int)(((ga >> 32) & 0x1FFFFFFu) | (2u << 30)) };
        // D# group1: data_size=2(4B) | tensor_dim0=16 | tensor_dim1=65536 |
        //            tile_dim0=16 | tile_dim1=16 | tensor_dim0_stride=16
        i32x8 g1 = { (int)0x00020000,        // workgroup_mask=0, data_size=2
                     (int)(16u << 16),       // tensor_dim0[15:0] @ bits[63:48]
                     0,                      // tensor_dim0 hi, tensor_dim1 lo
                     (int)((16u << 16) | 1u),// tensor_dim1 hi=1, tile_dim0=16
                     16,                     // tile_dim1=16, tile_dim2=0
                     16,                     // tensor_dim0_stride lo
                     0, 0 };
        i32x4 gz4 = { 0, 0, 0, 0 };
        i32x8 gz8 = { 0, 0, 0, 0, 0, 0, 0, 0 };
        __builtin_amdgcn_tensor_load_to_lds(g0, g1, gz4, gz4, gz8, 0);
    }

    // ---- prefetch f16 weights into cache while TDM runs ----
    __builtin_prefetch((const char*)w2bh + tid * 128, 0, 3);
    __builtin_prefetch((const char*)fwh + tid * 256, 0, 3);

    // ---- phase 0: stage small constants / per-point data ----
    if (tid < DD) {
        float s = g2b[tid] * rsqrtf(v2b[tid] + 1e-5f);
        sh_s2b[tid] = s;
        sh_o2b[tid] = b2b[tid] - m2b[tid] * s;
        sh_fb[tid]  = fb[tid];
        sh_c1b[tid] = c1b[tid];
    }
    for (int i = tid; i < DD * CIN; i += TPB) {
        sh_w2a[i] = w2a[i];
        sh_c1w[i] = c1w[i];
    }
    if (tid < CIN) {
        float s = g2a[tid] * rsqrtf(v2a[tid] + 1e-5f);
        sh_s2a[tid] = s;
        sh_o2a[tid] = b2a[tid] - m2a[tid] * s;
    }
    if (tid < 16) {
        int g = base + tid;
        sh_pix[tid] = py[g] * WW + px[g];
#pragma unroll
        for (int c = 0; c < CIN; ++c) sh_xc[tid * CIN + c] = xg[g * CIN + c];
    }
    if (tid < 32) __builtin_amdgcn_s_wait_tensorcnt(0);
    __syncthreads();

    // ---- per-thread hoisted state for the h1 pipeline ----
    const int hp  = tid >> 4;          // column (point) 0..15
    const int hd0 = (tid & 15) * 8;    // 8 consecutive d's
    float wr[8][CIN], s2bv[8], o2bv[8], c1wv[8][CIN], c1bv[8];
#pragma unroll
    for (int r = 0; r < 8; ++r) {
        int d = hd0 + r;
        s2bv[r] = sh_s2b[d];
        o2bv[r] = sh_o2b[d];
        c1bv[r] = sh_c1b[d];
#pragma unroll
        for (int c = 0; c < CIN; ++c) {
            wr[r][c]   = sh_w2a[d * CIN + c];
            c1wv[r][c] = sh_c1w[d * CIN + c];
        }
    }

    // ---- phase 1: neighbor gathers + BN2a; point_emb into cat[0..127] ----
    if (tid < 16 * KNB) {
        int p = tid & 15, j = tid >> 4;
        int nbi = sh_nbr[p * 16 + j + 1];        // idx = nb[:,1:,:]
#pragma unroll
        for (int c = 0; c < CIN; ++c)
            sh_tall[tid * CIN + c] =
                (xg[nbi * CIN + c] - sh_xc[p * CIN + c]) * sh_s2a[c] + sh_o2a[c];
    }
    {
        v8h pk;
#pragma unroll
        for (int r = 0; r < 8; ++r) {
            float a = c1bv[r];
#pragma unroll
            for (int c = 0; c < CIN; ++c) a += c1wv[r][c] * sh_xc[hp * CIN + c];
            pk[r] = (_Float16)a;
        }
        *(v8h*)&sh_cat[hp * 256 + hd0] = pk;
    }

    // ---- WMMA setup ----
    const int  lane = tid & 31;
    const int  wv   = tid >> 5;
    const int  l    = lane & 15;
    const bool hiL  = lane >= 16;
    const int  m0   = wv * 16;

    v16h afrag[4];
#pragma unroll
    for (int kk = 0; kk < 4; ++kk)
        afrag[kk] = load_a_frag(w2bh + (m0 + l) * DD + kk * 32 + (hiL ? 16 : 0));

    v8f vmax;
#pragma unroll
    for (int r = 0; r < 8; ++r) vmax[r] = -3.0e38f;

    __syncthreads();

    // ---- stage 2: h1 (VALU) -> WMMA -> running max; one barrier/iter ----
    for (int j = 0; j < KNB; ++j) {
        _Float16* hbuf = sh_h1[j & 1];
        {
            const float* t = &sh_tall[(j * 16 + hp) * CIN];
            float t0 = t[0], t1 = t[1], t2 = t[2], t3 = t[3], t4 = t[4];
            v8h pk;
#pragma unroll
            for (int r = 0; r < 8; ++r) {
                float a = wr[r][0] * t0 + wr[r][1] * t1 + wr[r][2] * t2 +
                          wr[r][3] * t3 + wr[r][4] * t4;
                a = a * s2bv[r] + o2bv[r];
                a = gelu_exact(a);
                pk[r] = (_Float16)a;
            }
            *(v8h*)&hbuf[hp * DD + hd0] = pk;
        }
        __syncthreads();

        v8f acc = {0.f, 0.f, 0.f, 0.f, 0.f, 0.f, 0.f, 0.f};
#pragma unroll
        for (int kk = 0; kk < 4; ++kk) {
            const v16h b = *(const v16h*)&hbuf[l * DD + kk * 32 + (hiL ? 16 : 0)];
            acc = __builtin_amdgcn_wmma_f32_16x16x32_f16(
                false, afrag[kk], false, b, (short)0, acc, false, false);
        }
#pragma unroll
        for (int r = 0; r < 8; ++r) vmax[r] = fmaxf(vmax[r], acc[r]);
    }

    // ---- neigh_emb into cat[128..255] (C/D layout: N=l, M=m0+r+hi*8) ----
#pragma unroll
    for (int r = 0; r < 8; ++r) {
        int m = m0 + r + (hiL ? 8 : 0);
        sh_cat[l * 256 + DD + m] = (_Float16)vmax[r];
    }
    __syncthreads();

    // ---- stage 3: feats = final_w @ concat, K=256 in 8 WMMA steps ----
    v8f f = {0.f, 0.f, 0.f, 0.f, 0.f, 0.f, 0.f, 0.f};
#pragma unroll
    for (int kk = 0; kk < 8; ++kk) {
        v16h a = load_a_frag(fwh + (m0 + l) * 256 + kk * 32 + (hiL ? 16 : 0));
        const v16h b = *(const v16h*)&sh_cat[l * 256 + kk * 32 + (hiL ? 16 : 0)];
        f = __builtin_amdgcn_wmma_f32_16x16x32_f16(
            false, a, false, b, (short)0, f, false, false);
    }

    // ---- bias + scatter into projection image ----
    const int pix = sh_pix[l];
#pragma unroll
    for (int r = 0; r < 8; ++r) {
        int m = m0 + r + (hiL ? 8 : 0);
        out[(size_t)m * HWSZ + pix] = f[r] + sh_fb[m];
    }
}

// ---------------------------------------------------------------------------
extern "C" void kernel_launch(void* const* d_in, const int* in_sizes, int n_in,
                              void* d_out, int out_size, void* d_ws, size_t ws_size,
                              hipStream_t stream) {
    const float* pc        = (const float*)d_in[0];
    const int*   neighbors = (const int*)  d_in[1];
    const int*   px        = (const int*)  d_in[2];
    const int*   py        = (const int*)  d_in[3];
    const float* ng        = (const float*)d_in[4];
    const float* nb        = (const float*)d_in[5];
    const float* nm        = (const float*)d_in[6];
    const float* nv        = (const float*)d_in[7];
    const float* c1w       = (const float*)d_in[8];
    const float* c1b       = (const float*)d_in[9];
    const float* g2a       = (const float*)d_in[10];
    const float* b2a       = (const float*)d_in[11];
    const float* m2a       = (const float*)d_in[12];
    const float* v2a       = (const float*)d_in[13];
    const float* w2a       = (const float*)d_in[14];
    const float* g2b       = (const float*)d_in[15];
    const float* b2b       = (const float*)d_in[16];
    const float* m2b       = (const float*)d_in[17];
    const float* v2b       = (const float*)d_in[18];
    const float* w2b       = (const float*)d_in[19];
    const float* fw        = (const float*)d_in[20];
    const float* fbias     = (const float*)d_in[21];

    float*     xws  = (float*)d_ws;                                  // N*5 f32
    _Float16*  w2bh = (_Float16*)((char*)d_ws + (size_t)NPTS * CIN * 4);
    _Float16*  fwh  = w2bh + DD * DD;
    float*     out  = (float*)d_out;

    const int outN = DD * HH * WW;
    init_out_kernel<<<(outN + TPB - 1) / TPB, TPB, 0, stream>>>(out, outN);
    prep_kernel<<<(NPTS + TPB - 1) / TPB, TPB, 0, stream>>>(
        pc, ng, nb, nm, nv, w2b, fw, xws, w2bh, fwh);
    fused_embed_kernel<<<NPTS / 16, TPB, 0, stream>>>(
        xws, neighbors, px, py, c1w, c1b,
        g2a, b2a, m2a, v2a, w2a,
        g2b, b2b, m2b, v2b,
        w2bh, fwh, fbias, out);
}
